// SparseAutoencoder_33440615367235
// MI455X (gfx1250) — compile-verified, matrix operand order checked by reading, not run on hardware
//
#include <hip/hip_runtime.h>
#include <hip/hip_bf16.h>

// ---------------------------------------------------------------------------
// Ext-vector types (trivially constructible -> legal in unions).
// ---------------------------------------------------------------------------
typedef __attribute__((ext_vector_type(8)))  float        v8f;
typedef __attribute__((ext_vector_type(16))) __bf16       v16bf;
typedef __attribute__((ext_vector_type(4)))  unsigned int v4u;
typedef __attribute__((ext_vector_type(8)))  unsigned int v8u;

union FragBF16 { v8u u; v16bf v; };

__device__ __forceinline__ unsigned short f32_to_bf16_rne(float f) {
  unsigned u = __float_as_uint(f);
  unsigned r = (u + 0x7FFFu + ((u >> 16) & 1u)) >> 16;
  return (unsigned short)r;
}

__device__ __forceinline__ unsigned fkey(float f) {
  unsigned u = __float_as_uint(f);
  return (u & 0x80000000u) ? ~u : (u | 0x80000000u);
}

// Generic (flat) pointer to LDS byte offset: LDS aperture keeps the workgroup-
// relative offset in addr[31:0] (ISA 10.2 aperture mapping).
__device__ __forceinline__ unsigned lds_off(const void* p) {
  return (unsigned)(size_t)p;
}

// CDNA5 async DMA: 16 B global -> LDS per lane, tracked by ASYNCcnt.
__device__ __forceinline__ void async_copy16(unsigned lds_byte_off, const void* g) {
  asm volatile("global_load_async_to_lds_b128 %0, %1, off"
               :: "v"(lds_byte_off), "v"(g) : "memory");
}
#define WAIT_ASYNC_0() asm volatile("s_wait_asynccnt 0x0" ::: "memory")
#define WAIT_ASYNC_6() asm volatile("s_wait_asynccnt 0x6" ::: "memory")

// ---------------------------------------------------------------------------
// Kernel 1: f32 -> bf16 conversion (8 elems / thread, 16B stores)
// ---------------------------------------------------------------------------
__global__ __launch_bounds__(256) void sae_cvt_bf16(
    const float* __restrict__ src, unsigned short* __restrict__ dst, long n) {
  long i = ((long)blockIdx.x * blockDim.x + threadIdx.x) * 8;
  if (i >= n) return;
  uint4 o;
  o.x = (unsigned)f32_to_bf16_rne(src[i + 0]) | ((unsigned)f32_to_bf16_rne(src[i + 1]) << 16);
  o.y = (unsigned)f32_to_bf16_rne(src[i + 2]) | ((unsigned)f32_to_bf16_rne(src[i + 3]) << 16);
  o.z = (unsigned)f32_to_bf16_rne(src[i + 4]) | ((unsigned)f32_to_bf16_rne(src[i + 5]) << 16);
  o.w = (unsigned)f32_to_bf16_rne(src[i + 6]) | ((unsigned)f32_to_bf16_rne(src[i + 7]) << 16);
  *(uint4*)(dst + i) = o;
}

// ---------------------------------------------------------------------------
// Kernel 2: encode GEMM  H[B,N] = Xb[B,D] * Wb[N,D]^T
//   v_wmma_f32_16x16x32_bf16, 128x256 WG tile, BK=32, 8 wave32 waves (2x4),
//   64x64 per wave (16 accumulators -> 16 WMMA : 16 ds_load_b128 per K-step).
//   Double-buffered LDS filled by global_load_async_to_lds_b128 (ASYNCcnt).
// ---------------------------------------------------------------------------
#define BM 128
#define BN 256
#define BK 32
#define LSTR 40                       // halfs per LDS row (32 data + 8 pad)
#define STAGE_HALFS ((BM + BN) * LSTR)
#define STAGE_BYTES (STAGE_HALFS * 2) // 30720 B per buffer

__global__ __launch_bounds__(256) void sae_encode_wmma(
    const unsigned short* __restrict__ Xb,  // [B,D] bf16
    const unsigned short* __restrict__ Wb,  // [N,D] bf16
    float* __restrict__ H,                  // [B,N]
    int B, int N, int D) {
  __shared__ __align__(16) unsigned short lds[2][STAGE_HALFS];  // ~60 KB

  const int tid  = threadIdx.x;
  const int lane = tid & 31;
  const int wave = tid >> 5;     // 0..7
  const int wy   = wave >> 2;    // 0..1 -> M offset wy*64
  const int wx   = wave & 3;     // 0..3 -> N offset wx*64
  const int mBase = blockIdx.y * BM;
  const int nBase = blockIdx.x * BN;

  const int laneHalf = lane >> 4;    // 0 | 1
  const int lr       = lane & 15;
  const int koff0    = laneHalf * 8; // ISA 16-bit operand K layout

  // --- per-thread DMA chunk assignment (16 B each) -------------------------
  // A tile: 128 rows x 4 chunks = 512 -> 2/thread. B tile: 256x4 = 1024 -> 4/thread.
  const int aR0 = tid >> 2,          aC0 = (tid & 3) * 8;
  const int aR1 = (tid + 256) >> 2,  aC1 = ((tid + 256) & 3) * 8;
  const unsigned short* gA0 = Xb + (size_t)(mBase + aR0) * D + aC0;
  const unsigned short* gA1 = Xb + (size_t)(mBase + aR1) * D + aC1;
  const unsigned lA0 = (unsigned)(aR0 * LSTR + aC0) * 2;
  const unsigned lA1 = (unsigned)(aR1 * LSTR + aC1) * 2;

  const unsigned short* gB[4];
  unsigned lB[4];
#pragma unroll
  for (int j = 0; j < 4; ++j) {
    int c  = tid + j * 256;
    int bR = c >> 2, bC = (c & 3) * 8;
    gB[j]  = Wb + (size_t)(nBase + bR) * D + bC;
    lB[j]  = (unsigned)((BM * LSTR) + bR * LSTR + bC) * 2;
  }
  const unsigned ldsBase = lds_off(&lds[0][0]);

  v8f acc[4][4];
#pragma unroll
  for (int i = 0; i < 4; ++i)
#pragma unroll
    for (int j = 0; j < 4; ++j)
#pragma unroll
      for (int r = 0; r < 8; ++r) acc[i][j][r] = 0.0f;

  const int nIter = D / BK;  // 24

  // Prologue: DMA stage 0 into buffer 0.
  {
    const unsigned so = ldsBase;
    async_copy16(so + lA0, gA0);
    async_copy16(so + lA1, gA1);
#pragma unroll
    for (int j = 0; j < 4; ++j) async_copy16(so + lB[j], gB[j]);
  }

  for (int it = 0; it < nIter; ++it) {
    const int cur = it & 1;
    if (it + 1 < nIter) {
      // Prefetch stage it+1 into the other buffer (last read 2 stages ago,
      // protected by the trailing barrier of iteration it-1).
      const unsigned so = ldsBase + (unsigned)((it + 1) & 1) * STAGE_BYTES;
      const size_t  ko = (size_t)(it + 1) * BK;
      async_copy16(so + lA0, gA0 + ko);
      async_copy16(so + lA1, gA1 + ko);
#pragma unroll
      for (int j = 0; j < 4; ++j) async_copy16(so + lB[j], gB[j] + ko);
      WAIT_ASYNC_6();   // retire my stage-`it` DMAs (oldest 6 of <=12)
    } else {
      WAIT_ASYNC_0();
    }
    __syncthreads();    // whole tile `it` resident in lds[cur]

    FragBF16 a[4], b[4];
#pragma unroll
    for (int fm = 0; fm < 4; ++fm) {
      int r = wy * 64 + fm * 16 + lr;
      v4u t0 = *(const v4u*)&lds[cur][r * LSTR + koff0];
      v4u t1 = *(const v4u*)&lds[cur][r * LSTR + koff0 + 16];
#pragma unroll
      for (int j = 0; j < 4; ++j) { a[fm].u[j] = t0[j]; a[fm].u[4 + j] = t1[j]; }
    }
#pragma unroll
    for (int fn = 0; fn < 4; ++fn) {
      int r = wx * 64 + fn * 16 + lr;   // B column n == W row n (K contiguous)
      v4u t0 = *(const v4u*)&lds[cur][BM * LSTR + r * LSTR + koff0];
      v4u t1 = *(const v4u*)&lds[cur][BM * LSTR + r * LSTR + koff0 + 16];
#pragma unroll
      for (int j = 0; j < 4; ++j) { b[fn].u[j] = t0[j]; b[fn].u[4 + j] = t1[j]; }
    }

#pragma unroll
    for (int fm = 0; fm < 4; ++fm)
#pragma unroll
      for (int fn = 0; fn < 4; ++fn)
        acc[fm][fn] = __builtin_amdgcn_wmma_f32_16x16x32_bf16(
            false, a[fm].v, false, b[fn].v, (short)0, acc[fm][fn], false, false);

    __syncthreads();    // all waves done reading lds[cur] before it is re-DMAed
  }

  // Epilogue: VGPR r holds M = r (lanes 0-15) / r+8 (lanes 16-31), N = lane&15.
#pragma unroll
  for (int fm = 0; fm < 4; ++fm) {
    int rb = mBase + wy * 64 + fm * 16 + laneHalf * 8;
#pragma unroll
    for (int fn = 0; fn < 4; ++fn) {
      int cb = nBase + wx * 64 + fn * 16 + lr;
#pragma unroll
      for (int r = 0; r < 8; ++r)
        H[(size_t)(rb + r) * N + cb] = acc[fm][fn][r];
    }
  }
}

// ---------------------------------------------------------------------------
// Kernel 3: per-row exact top-k via LDS-resident radix select (320KB LDS/WGP).
//   Row (128 KB) DMAed once into LDS via async-to-LDS; 4 histogram passes +
//   selection pass all read LDS. Writes mask, zeroes non-topk H, emits lists.
// ---------------------------------------------------------------------------
__global__ __launch_bounds__(256) void sae_topk_mask(
    float* __restrict__ H,     // [B,N] in/out (sparse_h slot)
    float* __restrict__ Mask,  // [B,N]
    int*   __restrict__ outIdx,   // [B,K]
    float* __restrict__ outVal,   // [B,K]
    int N, int K) {
  extern __shared__ float smem[];
  float*    rowv = smem;                       // N floats
  unsigned* hist = (unsigned*)(smem + N);      // 256 bins
  __shared__ unsigned selDigit, kRemS, tieTicket, listCnt;

  const int row = blockIdx.x;
  const int tid = threadIdx.x;
  float* hRow = H + (size_t)row * N;
  float* mRow = Mask + (size_t)row * N;

  {
    const unsigned rb = lds_off(rowv);
    for (int i = tid * 4; i < N; i += 256 * 4)
      async_copy16(rb + (unsigned)i * 4, hRow + i);
    WAIT_ASYNC_0();
  }
  if (tid == 0) { kRemS = (unsigned)K; tieTicket = 0; listCnt = 0; }
  __syncthreads();

  unsigned prefix = 0, pmask = 0;
  for (int shift = 24; shift >= 0; shift -= 8) {
    hist[tid] = 0;
    __syncthreads();
    for (int i = tid; i < N; i += 256) {
      unsigned u = fkey(rowv[i]);
      if ((u & pmask) == prefix) atomicAdd(&hist[(u >> shift) & 255u], 1u);
    }
    __syncthreads();
    if (tid == 0) {
      unsigned kr = kRemS;
      unsigned d  = 255;
      for (;; --d) {
        unsigned c = hist[d];
        if (kr <= c || d == 0) break;
        kr -= c;
      }
      selDigit = d;
      kRemS    = kr;  // survivors to take inside selected bin
    }
    __syncthreads();
    prefix |= (selDigit << shift);
    pmask  |= (255u << shift);
    __syncthreads();
  }

  const unsigned T        = prefix;  // exact key of the k-th largest element
  const unsigned tieQuota = kRemS;   // how many key==T entries to accept

  for (int i = tid; i < N; i += 256) {
    float f    = rowv[i];
    unsigned u = fkey(f);
    bool sel = false;
    if (u > T) sel = true;
    else if (u == T) sel = (atomicAdd(&tieTicket, 1u) < tieQuota);
    mRow[i] = sel ? 1.0f : 0.0f;
    hRow[i] = sel ? f : 0.0f;
    if (sel) {
      unsigned p = atomicAdd(&listCnt, 1u);
      if (p < (unsigned)K) {
        outIdx[(size_t)row * K + p] = i;
        outVal[(size_t)row * K + p] = f;
      }
    }
  }
}

// ---------------------------------------------------------------------------
// Kernel 4: sparse decode  recon[b,:] = sum_j val_j * W[idx_j,:]
//   W (100 MB f32) is L2-resident on MI455X (192 MB L2) -> gathers hit L2.
// ---------------------------------------------------------------------------
__global__ __launch_bounds__(256) void sae_decode(
    const float* __restrict__ W, const int* __restrict__ idxL,
    const float* __restrict__ valL, float* __restrict__ recon,
    int D, int N, int K) {
  __shared__ int   sIdx[64];
  __shared__ float sVal[64];
  const int row = blockIdx.x;
  const int tid = threadIdx.x;
  if (tid < K) {
    sIdx[tid] = idxL[(size_t)row * K + tid] & (N - 1);  // N is pow2; safety clamp
    sVal[tid] = valL[(size_t)row * K + tid];
  }
  __syncthreads();
  for (int d = tid; d < D; d += 256) {
    float acc = 0.0f;
#pragma unroll 4
    for (int j = 0; j < K; ++j) acc += sVal[j] * W[(size_t)sIdx[j] * D + d];
    recon[(size_t)row * D + d] = acc;
  }
}

// ---------------------------------------------------------------------------
// Launch: convert -> encode (WMMA + async DMA) -> topk/mask -> decode
// ---------------------------------------------------------------------------
extern "C" void kernel_launch(void* const* d_in, const int* in_sizes, int n_in,
                              void* d_out, int out_size, void* d_ws, size_t ws_size,
                              hipStream_t stream) {
  const float* x = (const float*)d_in[0];  // [B,D]
  const float* W = (const float*)d_in[1];  // [N,D]
  const int D = 768;
  const int B = in_sizes[0] / D;   // 8192
  const int N = in_sizes[1] / D;   // 32768
  const int K = 64;

  float* recon = (float*)d_out;                     // [B,D]
  float* hOut  = recon + (size_t)B * D;             // [B,N] sparse_h slot
  float* mask  = hOut + (size_t)B * N;              // [B,N]

  char* ws = (char*)d_ws;
  unsigned short* Xb = (unsigned short*)ws;  ws += (size_t)B * D * 2;
  unsigned short* Wb = (unsigned short*)ws;  ws += (size_t)N * D * 2;
  int*   idxL = (int*)ws;                    ws += (size_t)B * K * 4;
  float* valL = (float*)ws;

  {
    long n = (long)B * D;
    sae_cvt_bf16<<<(unsigned)((n / 8 + 255) / 256), 256, 0, stream>>>(x, Xb, n);
    n = (long)N * D;
    sae_cvt_bf16<<<(unsigned)((n / 8 + 255) / 256), 256, 0, stream>>>(W, Wb, n);
  }

  dim3 g(N / BN, B / BM);  // (128, 64)
  sae_encode_wmma<<<g, 256, 0, stream>>>(Xb, Wb, hOut, B, N, D);

  size_t shmem = (size_t)N * sizeof(float) + 256 * sizeof(unsigned);
  (void)hipFuncSetAttribute((const void*)sae_topk_mask,
                            hipFuncAttributeMaxDynamicSharedMemorySize, (int)shmem);
  sae_topk_mask<<<B, 256, shmem, stream>>>(hOut, mask, idxL, valL, N, K);

  sae_decode<<<B, 256, 0, stream>>>(W, idxL, valL, recon, D, N, K);
}
